// Generator_20177756357012
// MI455X (gfx1250) — matrix operand order repaired and checked by reading, not compiled, not run on hardware
//
#include <hip/hip_runtime.h>

// ---------------------------------------------------------------------------
// LSTM scan (B=256, E=1024, H=1024, 128 steps) for gfx1250 / MI455X.
// Compute path: V_WMMA_F32_16X16X32_BF16 (bf16 A/B, fp32 accumulate).
// Weights are L2-resident (18 MB bf16 << 192 MB L2) -> pure matrix-core bound,
// so we maximize WMMA issue density: 2x M register blocking amortizes the
// gate-weight (B) fragment loads, and biases ride in the WMMA C-operand init.
// ---------------------------------------------------------------------------

typedef __attribute__((ext_vector_type(16))) __bf16 v16bf;
typedef __attribute__((ext_vector_type(8)))  float  v8f;

#define SEQ_LEN 128
#define BDIM    256     // batch
#define EDIM    1024    // embedding
#define HDIM    1024    // hidden
#define G4H     4096    // 4*H
#define KTOT    2048    // E + H (fused [x|h] K dimension)

// fp32 -> bf16 round-to-nearest-even
static __device__ __forceinline__ unsigned short f2bf(float f) {
    unsigned int u = __float_as_uint(f);
    unsigned int r = 0x7FFFu + ((u >> 16) & 1u);
    return (unsigned short)((u + r) >> 16);
}

static __device__ __forceinline__ float sigmoidf_(float x) {
    return 1.0f / (1.0f + __expf(-x));
}

static __device__ __forceinline__ v8f splat8(float x) {
    v8f r = {x, x, x, x, x, x, x, x};
    return r;
}

union FragBF { v16bf v; int4 q[2]; };

// ---------------------------------------------------------------------------
// One-time per-launch weight prep:
//  Wcomb[n][k] = bf16( k < E ? W_ih[n][k] : W_hh[n][k-E] )   (4096 x 2048)
//  WfcB        = bf16(W_fc)                                  (1024 x 1024)
//  bcomb[n]    = b_ih[n] + b_hh[n]                           (4096)
// ---------------------------------------------------------------------------
__global__ void prep_weights_kernel(const float* __restrict__ Wih,
                                    const float* __restrict__ Whh,
                                    const float* __restrict__ bih,
                                    const float* __restrict__ bhh,
                                    const float* __restrict__ Wfc,
                                    unsigned short* __restrict__ Wcomb,
                                    unsigned short* __restrict__ WfcB,
                                    float* __restrict__ bcomb) {
    const size_t NW = (size_t)G4H * KTOT;           // 8388608
    const size_t NF = (size_t)EDIM * HDIM;          // 1048576
    const size_t total = NW + NF + G4H;
    const size_t stride = (size_t)gridDim.x * blockDim.x;
    for (size_t i = (size_t)blockIdx.x * blockDim.x + threadIdx.x; i < total; i += stride) {
        if (i < NW) {
            size_t n = i / KTOT, k = i - n * KTOT;
            float w = (k < EDIM) ? Wih[n * EDIM + k] : Whh[n * HDIM + (k - EDIM)];
            Wcomb[i] = f2bf(w);
        } else if (i < NW + NF) {
            size_t j = i - NW;
            WfcB[j] = f2bf(Wfc[j]);
        } else {
            size_t n = i - NW - NF;
            bcomb[n] = bih[n] + bhh[n];
        }
    }
}

// ---------------------------------------------------------------------------
// Per-launch state init: xh0 = bf16([start_emb | h0]), cbuf = c0 (fp32).
// ---------------------------------------------------------------------------
__global__ void init_state_kernel(const float* __restrict__ start_emb,
                                  const float* __restrict__ h0,
                                  const float* __restrict__ c0,
                                  unsigned short* __restrict__ xh0,
                                  float* __restrict__ cbuf) {
    const size_t NXH = (size_t)BDIM * KTOT;         // 524288
    const size_t NC  = (size_t)BDIM * HDIM;         // 262144
    const size_t total = NXH + NC;
    const size_t stride = (size_t)gridDim.x * blockDim.x;
    for (size_t i = (size_t)blockIdx.x * blockDim.x + threadIdx.x; i < total; i += stride) {
        if (i < NXH) {
            size_t m = i / KTOT, k = i - m * KTOT;
            float v = (k < EDIM) ? start_emb[m * EDIM + k] : h0[m * HDIM + (k - EDIM)];
            xh0[i] = f2bf(v);
        } else {
            size_t j = i - NXH;
            cbuf[j] = c0[j];
        }
    }
}

// ---------------------------------------------------------------------------
// Fused gates GEMM + LSTM cell.
// Grid: (HDIM/64, BDIM/32), 128 threads = 4 waves; wave w owns column tile
// n0 = bx*64 + w*16 and BOTH 16-row subtiles of the block's 32-row M tile.
// Per wave: 4 gates x 2 M-subtiles = 8 fp32 WMMA accumulators, initialized
// to the (column-uniform) gate bias. Per K-step: 4 A loads + 8 B loads ->
// 8 WMMAs (1.5 b128 loads per WMMA).
// A frag (bf16 16x32): lane = row, lanes<16: K 0..15; >=16: 16..31.
// B frag (bf16 32x16): lane = out-col = W row, lanes<16: K 0..15; >=16: 16..31.
// Both are contiguous b128 loads from row-major (K-contiguous) storage.
// ---------------------------------------------------------------------------
__global__ __launch_bounds__(128)
void lstm_step_kernel(const unsigned short* __restrict__ xh,      // [B][KTOT] bf16
                      const unsigned short* __restrict__ Wcomb,   // [4H][KTOT] bf16
                      const float* __restrict__ bcomb,            // [4H]
                      float* __restrict__ cbuf,                   // [B][H] in/out (fp32)
                      float* __restrict__ hout,                   // [B][H] out (fp32)
                      unsigned short* __restrict__ xh_next) {     // [B][KTOT]; writes h half
    const int wave = threadIdx.x >> 5;               // 0..3 -> column subtile
    const int lane = threadIdx.x & 31;
    const int n0   = blockIdx.x * 64 + wave * 16;    // h-column tile base
    const int m0   = blockIdx.y * 32;                // batch-row tile base

    const int hi  = lane >> 4;                       // lane half
    const int l15 = lane & 15;
    const int kAo = hi * 16;                         // A-frag K sub-offset
    const int kBo = hi * 16;                         // B-frag K sub-offset

    const unsigned short* aB0 = xh + (size_t)(m0 + l15) * KTOT;
    const unsigned short* aB1 = xh + (size_t)(m0 + 16 + l15) * KTOT;
    const unsigned short* bB0 = Wcomb + (size_t)(0 * HDIM + n0 + l15) * KTOT;
    const unsigned short* bB1 = Wcomb + (size_t)(1 * HDIM + n0 + l15) * KTOT;
    const unsigned short* bB2 = Wcomb + (size_t)(2 * HDIM + n0 + l15) * KTOT;
    const unsigned short* bB3 = Wcomb + (size_t)(3 * HDIM + n0 + l15) * KTOT;

    const int col = n0 + l15;
    const float bi = bcomb[0 * HDIM + col];
    const float bf = bcomb[1 * HDIM + col];
    const float bg = bcomb[2 * HDIM + col];
    const float bo = bcomb[3 * HDIM + col];

    // acc[gate][m_subtile]; bias folded into the WMMA C operand (uniform per N-col).
    v8f a00 = splat8(bi), a01 = splat8(bi);
    v8f a10 = splat8(bf), a11 = splat8(bf);
    v8f a20 = splat8(bg), a21 = splat8(bg);
    v8f a30 = splat8(bo), a31 = splat8(bo);

    for (int k = 0; k < KTOT; k += 32) {
        FragBF fa0, fa1, fb;
        fa0.q[0] = *(const int4*)(aB0 + k + kAo);
        fa0.q[1] = *(const int4*)(aB0 + k + kAo + 8);
        fa1.q[0] = *(const int4*)(aB1 + k + kAo);
        fa1.q[1] = *(const int4*)(aB1 + k + kAo + 8);

        fb.q[0] = *(const int4*)(bB0 + k + kBo);
        fb.q[1] = *(const int4*)(bB0 + k + kBo + 8);
        a00 = __builtin_amdgcn_wmma_f32_16x16x32_bf16(false, fa0.v, false, fb.v, (short)0, a00, false, false);
        a01 = __builtin_amdgcn_wmma_f32_16x16x32_bf16(false, fa1.v, false, fb.v, (short)0, a01, false, false);

        fb.q[0] = *(const int4*)(bB1 + k + kBo);
        fb.q[1] = *(const int4*)(bB1 + k + kBo + 8);
        a10 = __builtin_amdgcn_wmma_f32_16x16x32_bf16(false, fa0.v, false, fb.v, (short)0, a10, false, false);
        a11 = __builtin_amdgcn_wmma_f32_16x16x32_bf16(false, fa1.v, false, fb.v, (short)0, a11, false, false);

        fb.q[0] = *(const int4*)(bB2 + k + kBo);
        fb.q[1] = *(const int4*)(bB2 + k + kBo + 8);
        a20 = __builtin_amdgcn_wmma_f32_16x16x32_bf16(false, fa0.v, false, fb.v, (short)0, a20, false, false);
        a21 = __builtin_amdgcn_wmma_f32_16x16x32_bf16(false, fa1.v, false, fb.v, (short)0, a21, false, false);

        fb.q[0] = *(const int4*)(bB3 + k + kBo);
        fb.q[1] = *(const int4*)(bB3 + k + kBo + 8);
        a30 = __builtin_amdgcn_wmma_f32_16x16x32_bf16(false, fa0.v, false, fb.v, (short)0, a30, false, false);
        a31 = __builtin_amdgcn_wmma_f32_16x16x32_bf16(false, fa1.v, false, fb.v, (short)0, a31, false, false);
    }

    // C/D layout: VGPR v, lanes<16 -> (M = v, N = lane); lanes>=16 -> (M = v+8).
#pragma unroll
    for (int r = 0; r < 2; ++r) {
        const v8f gi = r ? a01 : a00;
        const v8f gf = r ? a11 : a10;
        const v8f gg = r ? a21 : a20;
        const v8f go = r ? a31 : a30;
#pragma unroll
        for (int v = 0; v < 8; ++v) {
            const int row = m0 + 16 * r + v + hi * 8;
            const float iv = sigmoidf_(gi[v]);
            const float fv = sigmoidf_(gf[v]);
            const float gv = tanhf(gg[v]);
            const float ov = sigmoidf_(go[v]);
            const size_t idx = (size_t)row * HDIM + col;
            const float cn = fv * cbuf[idx] + iv * gv;
            const float hn = ov * tanhf(cn);
            cbuf[idx] = cn;                                      // fp32 cell state in place
            hout[idx] = hn;                                      // fp32 h (final answer)
            xh_next[(size_t)row * KTOT + EDIM + col] = f2bf(hn); // bf16 h for next GEMMs
        }
    }
}

// ---------------------------------------------------------------------------
// fc GEMM: x_next = h_next @ W_fc^T + b_fc -> bf16 x-half of xh_next.
// Grid: (EDIM/128, BDIM/32), 128 threads = 4 waves; wave owns a 32x32 output
// block (2 M-subtiles x 2 N-subtiles = 4 accumulators, bias in C-init).
// Per K-step: 4 A loads + 4 B loads -> 4 WMMAs (2.0 loads per WMMA).
// ---------------------------------------------------------------------------
__global__ __launch_bounds__(128)
void fc_step_kernel(unsigned short* __restrict__ xh_next,       // reads h half, writes x half
                    const unsigned short* __restrict__ WfcB,    // [E][H] bf16
                    const float* __restrict__ bfc) {            // [E]
    const int wave = threadIdx.x >> 5;               // 0..3
    const int lane = threadIdx.x & 31;
    const int n0   = blockIdx.x * 128 + wave * 32;   // output (E) column tile (32 wide)
    const int m0   = blockIdx.y * 32;

    const int hi  = lane >> 4;
    const int l15 = lane & 15;
    const int kAo = hi * 16;
    const int kBo = hi * 16;

    const unsigned short* aB0 = xh_next + (size_t)(m0 + l15) * KTOT + EDIM;       // h half
    const unsigned short* aB1 = xh_next + (size_t)(m0 + 16 + l15) * KTOT + EDIM;
    const unsigned short* bB0 = WfcB + (size_t)(n0 + l15) * HDIM;
    const unsigned short* bB1 = WfcB + (size_t)(n0 + 16 + l15) * HDIM;

    const int col0 = n0 + l15;
    const int col1 = n0 + 16 + l15;
    v8f a00 = splat8(bfc[col0]), a01 = splat8(bfc[col0]);   // [n_sub][m_sub]
    v8f a10 = splat8(bfc[col1]), a11 = splat8(bfc[col1]);

    for (int k = 0; k < HDIM; k += 32) {
        FragBF fa0, fa1, fb;
        fa0.q[0] = *(const int4*)(aB0 + k + kAo);
        fa0.q[1] = *(const int4*)(aB0 + k + kAo + 8);
        fa1.q[0] = *(const int4*)(aB1 + k + kAo);
        fa1.q[1] = *(const int4*)(aB1 + k + kAo + 8);

        fb.q[0] = *(const int4*)(bB0 + k + kBo);
        fb.q[1] = *(const int4*)(bB0 + k + kBo + 8);
        a00 = __builtin_amdgcn_wmma_f32_16x16x32_bf16(false, fa0.v, false, fb.v, (short)0, a00, false, false);
        a01 = __builtin_amdgcn_wmma_f32_16x16x32_bf16(false, fa1.v, false, fb.v, (short)0, a01, false, false);

        fb.q[0] = *(const int4*)(bB1 + k + kBo);
        fb.q[1] = *(const int4*)(bB1 + k + kBo + 8);
        a10 = __builtin_amdgcn_wmma_f32_16x16x32_bf16(false, fa0.v, false, fb.v, (short)0, a10, false, false);
        a11 = __builtin_amdgcn_wmma_f32_16x16x32_bf16(false, fa1.v, false, fb.v, (short)0, a11, false, false);
    }

#pragma unroll
    for (int v = 0; v < 8; ++v) {
        const int row0 = m0 + v + hi * 8;
        const int row1 = row0 + 16;
        xh_next[(size_t)row0 * KTOT + col0] = f2bf(a00[v]);
        xh_next[(size_t)row1 * KTOT + col0] = f2bf(a01[v]);
        xh_next[(size_t)row0 * KTOT + col1] = f2bf(a10[v]);
        xh_next[(size_t)row1 * KTOT + col1] = f2bf(a11[v]);
    }
}

// ---------------------------------------------------------------------------
// Launch: prep + init once per call (graph-safe, deterministic), then 128
// sequential (gates+cell, fc) kernel pairs ping-ponging the bf16 [x|h] buffer.
// d_out layout: [0, B*H) = h fp32, [B*H, 2*B*H) = c fp32 (c doubles as the
// working cell-state buffer, re-initialized from c0 every launch).
// ---------------------------------------------------------------------------
extern "C" void kernel_launch(void* const* d_in, const int* in_sizes, int n_in,
                              void* d_out, int out_size, void* d_ws, size_t ws_size,
                              hipStream_t stream) {
    const float* start_emb = (const float*)d_in[0];   // (B,1,E)
    const float* h0        = (const float*)d_in[1];   // (1,B,H)
    const float* c0        = (const float*)d_in[2];   // (1,B,H)
    const float* W_ih      = (const float*)d_in[3];   // (4H,E)
    const float* W_hh      = (const float*)d_in[4];   // (4H,H)
    const float* b_ih      = (const float*)d_in[5];   // (4H)
    const float* b_hh      = (const float*)d_in[6];   // (4H)
    const float* W_fc      = (const float*)d_in[7];   // (E,H)
    const float* b_fc      = (const float*)d_in[8];   // (E)

    float* hout = (float*)d_out;                      // [B][H]
    float* cbuf = hout + (size_t)BDIM * HDIM;         // [B][H]

    // Workspace layout (all 256B-aligned): ~21 MB total.
    char* ws = (char*)d_ws;
    unsigned short* Wcomb = (unsigned short*)ws;                          // 16 MB
    unsigned short* WfcB  = (unsigned short*)(ws + 16777216);             //  2 MB
    float*          bcomb = (float*)(ws + 16777216 + 2097152);            // 16 KB
    unsigned short* xh0   = (unsigned short*)(ws + 16777216 + 2097152 + 16384); // 1 MB
    unsigned short* xh1   = xh0 + (size_t)BDIM * KTOT;                    // 1 MB

    {   // weight prep: 9,441,280 elements
        const size_t total = (size_t)G4H * KTOT + (size_t)EDIM * HDIM + G4H;
        int blocks = (int)((total + 255) / 256);
        prep_weights_kernel<<<blocks, 256, 0, stream>>>(W_ih, W_hh, b_ih, b_hh, W_fc,
                                                        Wcomb, WfcB, bcomb);
    }
    {   // state init: 786,432 elements
        const size_t total = (size_t)BDIM * KTOT + (size_t)BDIM * HDIM;
        int blocks = (int)((total + 255) / 256);
        init_state_kernel<<<blocks, 256, 0, stream>>>(start_emb, h0, c0, xh0, cbuf);
    }

    unsigned short* xh[2] = {xh0, xh1};
    dim3 gridGates(HDIM / 64, BDIM / 32);  // (16, 8)
    dim3 gridFc(EDIM / 128, BDIM / 32);    // (8, 8)
    for (int t = 0; t < SEQ_LEN; ++t) {
        lstm_step_kernel<<<gridGates, 128, 0, stream>>>(xh[t & 1], Wcomb, bcomb,
                                                        cbuf, hout, xh[(t + 1) & 1]);
        if (t + 1 < SEQ_LEN) {
            fc_step_kernel<<<gridFc, 128, 0, stream>>>(xh[(t + 1) & 1], WfcB, b_fc);
        }
    }
}
